// SmDecoder_16312285790241
// MI455X (gfx1250) — matrix operand order repaired and checked by reading, not compiled, not run on hardware
//
#include <hip/hip_runtime.h>
#include <hip/hip_bf16.h>
#include <stdint.h>

// ---------------------------------------------------------------------------
// LSTM decoder for MI455X (gfx1250).
//  - bf16 WMMA (v_wmma_f32_16x16x32_bf16), f32 accumulate
//  - weights cast once to bf16; W_out (65MB) stays resident in the 192MB L2
//  - logits written with non-temporal stores so the 520MB output stream
//    does not evict W_out from L2
//  - gates GEMM + LSTM cell fused; h double-buffered -> 2 launches/step
//  - fully-unrolled ping-pong K pipeline, pinned with sched_barrier(0) so
//    each WMMA group overlaps the in-flight loads of the next fragment set
// ---------------------------------------------------------------------------

typedef __attribute__((ext_vector_type(16))) __bf16 v16bf;
typedef __attribute__((ext_vector_type(8)))  __bf16 v8bf;
typedef __attribute__((ext_vector_type(8)))  float  v8f;
typedef __attribute__((ext_vector_type(4)))  float  v4f;   // clang vector: OK for NT builtins

static constexpr int VOCAB = 32000, EMB = 512, HID = 1024, LAT = 512;
static constexpr int BATCH = 32, T = 128, TS = T - 1;     // 127 scan steps
static constexpr int NG = 4 * HID;                        // 4096 gate cols

#define SCHED_FENCE() __builtin_amdgcn_sched_barrier(0)

// ---- fp32 -> bf16 round-to-nearest-even -----------------------------------
__device__ __forceinline__ __bf16 f32_to_bf16(float f) {
  union { float f; uint32_t u; } v; v.f = f;
  uint32_t r = v.u + 0x7fffu + ((v.u >> 16) & 1u);
  union { uint16_t u; __bf16 h; } o; o.u = (uint16_t)(r >> 16);
  return o.h;
}

union ABFrag { v16bf v; v8bf h[2]; };

// A fragment (16x32 bf16). Row-major source, stride ld elems.
// Lanes 0-15: row M=lane, K 0..7 (v0-3) and 16..23 (v4-7);
// lanes 16-31: same rows, K 8..15 and 24..31.
__device__ __forceinline__ v16bf load_A(const __bf16* __restrict__ p, int ld, int lane) {
  const int row  = lane & 15;
  const int koff = (lane >> 4) << 3;            // 0 or 8
  const __bf16* q = p + row * ld + koff;
  ABFrag f;
  f.h[0] = *(const v8bf*)(q);
  f.h[1] = *(const v8bf*)(q + 16);
  return f.v;
}

// B fragment (32x16 bf16), B[k][n] = W[n0+n][k0+k], W row-major (ld).
// Lane l: column n=l&15; lanes 0-15 K 0..7 and 16..23, lanes 16-31 K 8..15 and 24..31.
__device__ __forceinline__ v16bf load_Bt(const __bf16* __restrict__ p, int ld, int lane) {
  const int n    = lane & 15;
  const int koff = (lane >> 4) << 3;            // 0 or 8
  const __bf16* q = p + n * ld + koff;
  ABFrag f;
  f.h[0] = *(const v8bf*)(q);
  f.h[1] = *(const v8bf*)(q + 16);
  return f.v;
}

#define WMMA_BF16(A, Bm, C) \
  __builtin_amdgcn_wmma_f32_16x16x32_bf16(false, (A), false, (Bm), (short)0, (C), false, false)

// ---------------------------------------------------------------------------
// Prep kernels
// ---------------------------------------------------------------------------
union BF4 { __bf16 b[4]; uint64_t u; };

// 4-wide cast; NT loads: the fp32 weights are read exactly once, keep them
// out of L2 so the bf16 copies (which we reuse 127x) stay resident.
__global__ void cast_bf16_kernel(const float* __restrict__ src,
                                 uint64_t* __restrict__ dst, int n4) {
  int i = blockIdx.x * blockDim.x + threadIdx.x;
  if (i >= n4) return;
  v4f v = __builtin_nontemporal_load((const v4f*)(src) + i);
  BF4 o;
  o.b[0] = f32_to_bf16(v.x); o.b[1] = f32_to_bf16(v.y);
  o.b[2] = f32_to_bf16(v.z); o.b[3] = f32_to_bf16(v.w);
  dst[i] = o.u;
}

// xbf[t][b][e] = bf16(emb[tok[b][t]][e]), t in [0,126]; 4-wide over e
__global__ void gather_x_kernel(const float* __restrict__ emb,
                                const int* __restrict__ tok,
                                uint64_t* __restrict__ xbf) {
  int i = blockIdx.x * blockDim.x + threadIdx.x;          // over TS*B*EMB/4
  if (i >= TS * BATCH * (EMB / 4)) return;
  int e4 = i % (EMB / 4);
  int b  = (i / (EMB / 4)) % BATCH;
  int t  = i / ((EMB / 4) * BATCH);
  int v  = tok[b * T + t];
  v4f x = __builtin_nontemporal_load((const v4f*)(emb + (size_t)v * EMB) + e4);
  BF4 o;
  o.b[0] = f32_to_bf16(x.x); o.b[1] = f32_to_bf16(x.y);
  o.b[2] = f32_to_bf16(x.z); o.b[3] = f32_to_bf16(x.w);
  xbf[i] = o.u;
}

// h0 = latent @ W_lh^T + b_lh  (32x1024, K=512); also c = 0
__global__ void init_h_kernel(const float* __restrict__ latent,
                              const float* __restrict__ Wlh,
                              const float* __restrict__ blh,
                              __bf16* __restrict__ hbf,
                              float* __restrict__ c) {
  int i = blockIdx.x * blockDim.x + threadIdx.x;
  if (i >= BATCH * HID) return;
  int j = i % HID, b = i / HID;
  const float* lv = latent + b * LAT;
  const float* w  = Wlh + j * LAT;
  float s = blh[j];
#pragma unroll 4
  for (int k = 0; k < LAT; ++k) s += lv[k] * w[k];
  hbf[i] = f32_to_bf16(s);
  c[i] = 0.0f;
}

// ---------------------------------------------------------------------------
// Fused gates GEMM + LSTM cell, fully-unrolled ping-pong pipeline.
// Wave (jt, mt) computes gate columns {g*1024 + j0..j0+15 | g=0..3} for batch
// rows mt*16..mt*16+15, so it holds i,f,g,o of its (b,j) block in registers
// and applies the cell update inline. h is double-buffered across steps.
// 128 waves total (64 j-tiles x 2 m-tiles).
// Unified K-step index s in [0,48): s<16 -> x/Wih stream, s>=16 -> h/Whh.
// ---------------------------------------------------------------------------
__global__ void __launch_bounds__(256)
gates_cell_wmma_kernel(const __bf16* __restrict__ xt,     // 32 x 512
                       const __bf16* __restrict__ hin,    // 32 x 1024 (read)
                       __bf16* __restrict__ hout,         // 32 x 1024 (write)
                       const __bf16* __restrict__ Wih,    // 4096 x 512
                       const __bf16* __restrict__ Whh,    // 4096 x 1024
                       const float*  __restrict__ bih,
                       const float*  __restrict__ bhh,
                       float* __restrict__ c) {           // 32 x 1024 (rmw)
  const int lane = threadIdx.x & 31;
  const int wid  = blockIdx.x * (blockDim.x >> 5) + (threadIdx.x >> 5); // 0..127
  const int jt   = wid >> 1;
  const int mt   = wid & 1;
  const int j0   = jt * 16;

  v8f acc[4];
#pragma unroll
  for (int g = 0; g < 4; ++g) acc[g] = (v8f){};

  const __bf16* xa = xt  + (size_t)mt * 16 * EMB;
  const __bf16* ha = hin + (size_t)mt * 16 * HID;
  const __bf16* wi[4];
  const __bf16* wh[4];
#pragma unroll
  for (int g = 0; g < 4; ++g) {
    wi[g] = Wih + (size_t)(g * HID + j0) * EMB;
    wh[g] = Whh + (size_t)(g * HID + j0) * HID;
  }

  // step s (constant after unroll): pick stream + offset
  auto ldA = [&](int s) -> v16bf {
    return (s < 16) ? load_A(xa + s * 32, EMB, lane)
                    : load_A(ha + (s - 16) * 32, HID, lane);
  };
  auto ldB = [&](int s, int g) -> v16bf {
    return (s < 16) ? load_Bt(wi[g] + s * 32, EMB, lane)
                    : load_Bt(wh[g] + (s - 16) * 32, HID, lane);
  };

  constexpr int NS = (EMB + HID) / 32;   // 48 K-steps total

  v16bf aE = ldA(0);
  v16bf bE[4];
#pragma unroll
  for (int g = 0; g < 4; ++g) bE[g] = ldB(0, g);

#pragma unroll
  for (int s = 0; s < NS; s += 2) {
    v16bf aO = ldA(s + 1);
    v16bf bO[4];
#pragma unroll
    for (int g = 0; g < 4; ++g) bO[g] = ldB(s + 1, g);
    SCHED_FENCE();   // odd-set loads stay above; even WMMAs overlap them
#pragma unroll
    for (int g = 0; g < 4; ++g) acc[g] = WMMA_BF16(aE, bE[g], acc[g]);
    if (s + 2 < NS) {
      aE = ldA(s + 2);
#pragma unroll
      for (int g = 0; g < 4; ++g) bE[g] = ldB(s + 2, g);
    }
    SCHED_FENCE();   // even-set loads stay above; odd WMMAs overlap them
#pragma unroll
    for (int g = 0; g < 4; ++g) acc[g] = WMMA_BF16(aO, bO[g], acc[g]);
  }

  // ---- cell update. C/D layout: value r of lane l is (m=r+8*(l>=16), n=l&15)
  const int j  = j0 + (lane & 15);
  const int mh = (lane >> 4) << 3;
  const float bi = bih[j]           + bhh[j];
  const float bf = bih[HID + j]     + bhh[HID + j];
  const float bg = bih[2 * HID + j] + bhh[2 * HID + j];
  const float bo = bih[3 * HID + j] + bhh[3 * HID + j];
#pragma unroll
  for (int r = 0; r < 8; ++r) {
    const int bb = mt * 16 + r + mh;
    const size_t idx = (size_t)bb * HID + j;
    float vi = acc[0][r] + bi;
    float vf = acc[1][r] + bf;
    float vg = acc[2][r] + bg;
    float vo = acc[3][r] + bo;
    float si = 1.0f / (1.0f + __expf(-vi));
    float sf = 1.0f / (1.0f + __expf(-vf));
    float so = 1.0f / (1.0f + __expf(-vo));
    float cc = sf * c[idx] + si * tanhf(vg);
    c[idx] = cc;
    hout[idx] = f32_to_bf16(so * tanhf(cc));
  }
}

// ---------------------------------------------------------------------------
// logits = h @ W_out^T + b_out, written to out[b][t][v] (fp32, NT stores).
// Each wave does 2 N-tiles (A fragments amortized over 4 WMMAs); 1000 waves.
// Fully-unrolled ping-pong pipeline (16 pairs of K-steps), fence-pinned.
// ---------------------------------------------------------------------------
__global__ void __launch_bounds__(256)
logits_wmma_kernel(const __bf16* __restrict__ hbf,   // 32 x 1024
                   const __bf16* __restrict__ Wout,  // 32000 x 1024
                   const float*  __restrict__ bout,
                   float* __restrict__ out) {        // + t*VOCAB, row stride TS*VOCAB
  const int lane = threadIdx.x & 31;
  const int wid  = blockIdx.x * (blockDim.x >> 5) + (threadIdx.x >> 5); // 0..999
  const int n0   = wid * 32;

  v8f acc00 = {}, acc01 = {}, acc10 = {}, acc11 = {};
  const __bf16* bp0 = Wout + (size_t)n0 * HID;
  const __bf16* bp1 = bp0 + (size_t)16 * HID;
  const __bf16* ha0 = hbf;
  const __bf16* ha1 = hbf + 16 * HID;

  constexpr int NS = HID / 32;           // 32 K-steps

  v16bf aE0 = load_A(ha0, HID, lane);
  v16bf aE1 = load_A(ha1, HID, lane);
  v16bf bE0 = load_Bt(bp0, HID, lane);
  v16bf bE1 = load_Bt(bp1, HID, lane);

#pragma unroll
  for (int s = 0; s < NS; s += 2) {
    const int kO = (s + 1) * 32;
    v16bf aO0 = load_A(ha0 + kO, HID, lane);
    v16bf aO1 = load_A(ha1 + kO, HID, lane);
    v16bf bO0 = load_Bt(bp0 + kO, HID, lane);
    v16bf bO1 = load_Bt(bp1 + kO, HID, lane);
    SCHED_FENCE();   // odd-set loads stay above; even WMMAs overlap them
    acc00 = WMMA_BF16(aE0, bE0, acc00);
    acc01 = WMMA_BF16(aE1, bE0, acc01);
    acc10 = WMMA_BF16(aE0, bE1, acc10);
    acc11 = WMMA_BF16(aE1, bE1, acc11);

    if (s + 2 < NS) {
      const int kE = (s + 2) * 32;
      aE0 = load_A(ha0 + kE, HID, lane);
      aE1 = load_A(ha1 + kE, HID, lane);
      bE0 = load_Bt(bp0 + kE, HID, lane);
      bE1 = load_Bt(bp1 + kE, HID, lane);
    }
    SCHED_FENCE();   // even-set loads stay above; odd WMMAs overlap them
    acc00 = WMMA_BF16(aO0, bO0, acc00);
    acc01 = WMMA_BF16(aO1, bO0, acc01);
    acc10 = WMMA_BF16(aO0, bO1, acc10);
    acc11 = WMMA_BF16(aO1, bO1, acc11);
  }

  const int nA = n0 + (lane & 15);
  const int nB = nA + 16;
  const float bsA = bout[nA];
  const float bsB = bout[nB];
  const int mh = (lane >> 4) << 3;
  const size_t rs = (size_t)TS * VOCAB;
#pragma unroll
  for (int r = 0; r < 8; ++r) {
    const int m = r + mh;
    __builtin_nontemporal_store(acc00[r] + bsA, out + (size_t)m * rs + nA);
    __builtin_nontemporal_store(acc10[r] + bsB, out + (size_t)m * rs + nB);
    __builtin_nontemporal_store(acc01[r] + bsA, out + (size_t)(m + 16) * rs + nA);
    __builtin_nontemporal_store(acc11[r] + bsB, out + (size_t)(m + 16) * rs + nB);
  }
}

// ---------------------------------------------------------------------------
// Host side
// ---------------------------------------------------------------------------
static inline size_t align256(size_t x) { return (x + 255) & ~(size_t)255; }

extern "C" void kernel_launch(void* const* d_in, const int* in_sizes, int n_in,
                              void* d_out, int out_size, void* d_ws, size_t ws_size,
                              hipStream_t stream) {
  (void)in_sizes; (void)n_in; (void)out_size; (void)ws_size;
  const float* latent = (const float*)d_in[0];
  const int*   tok    = (const int*)  d_in[1];
  const float* emb    = (const float*)d_in[2];
  const float* Wlh    = (const float*)d_in[3];
  const float* blh    = (const float*)d_in[4];
  const float* Wih    = (const float*)d_in[5];
  const float* bih    = (const float*)d_in[6];
  const float* Whh    = (const float*)d_in[7];
  const float* bhh    = (const float*)d_in[8];
  const float* Wout   = (const float*)d_in[9];
  const float* bout   = (const float*)d_in[10];
  float* out = (float*)d_out;

  // Workspace carve-up (~83 MB)
  char* ws = (char*)d_ws;
  size_t off = 0;
  __bf16* Wih_bf  = (__bf16*)(ws + off); off += align256((size_t)NG * EMB * 2);
  __bf16* Whh_bf  = (__bf16*)(ws + off); off += align256((size_t)NG * HID * 2);
  __bf16* Wout_bf = (__bf16*)(ws + off); off += align256((size_t)VOCAB * HID * 2);
  __bf16* x_bf    = (__bf16*)(ws + off); off += align256((size_t)TS * BATCH * EMB * 2);
  __bf16* h_bf0   = (__bf16*)(ws + off); off += align256((size_t)BATCH * HID * 2);
  __bf16* h_bf1   = (__bf16*)(ws + off); off += align256((size_t)BATCH * HID * 2);
  float*  c_st    = (float*) (ws + off); off += align256((size_t)BATCH * HID * 4);
  __bf16* h_buf[2] = { h_bf0, h_bf1 };

  const int TB = 256;
  // 1) weight casts fp32 -> bf16 (one pass; thereafter bf16 copies live in L2)
  {
    int n4 = NG * EMB / 4;
    cast_bf16_kernel<<<(n4 + TB - 1) / TB, TB, 0, stream>>>(Wih, (uint64_t*)Wih_bf, n4);
    n4 = NG * HID / 4;
    cast_bf16_kernel<<<(n4 + TB - 1) / TB, TB, 0, stream>>>(Whh, (uint64_t*)Whh_bf, n4);
    n4 = VOCAB * HID / 4;
    cast_bf16_kernel<<<(n4 + TB - 1) / TB, TB, 0, stream>>>(Wout, (uint64_t*)Wout_bf, n4);
  }
  // 2) embedding gather + cast for all 127 steps
  {
    int n4 = TS * BATCH * EMB / 4;
    gather_x_kernel<<<(n4 + TB - 1) / TB, TB, 0, stream>>>(emb, tok, (uint64_t*)x_bf);
  }
  // 3) h0 = latent @ W_lh^T + b_lh ; c = 0
  {
    int n = BATCH * HID;
    init_h_kernel<<<(n + TB - 1) / TB, TB, 0, stream>>>(latent, Wlh, blh, h_bf0, c_st);
  }

  // 4) sequential scan: 127 steps, 2 launches per step
  const int gatesGrid  = (2 * (HID / 16)) / 8;  // 128 waves / 8 per block = 16 blocks
  const int logitsGrid = (VOCAB / 32) / 8;      // 1000 waves / 8 per block = 125 blocks

  for (int t = 0; t < TS; ++t) {
    const __bf16* xt  = x_bf + (size_t)t * BATCH * EMB;
    const __bf16* hin = h_buf[t & 1];
    __bf16*       hot = h_buf[(t + 1) & 1];
    gates_cell_wmma_kernel<<<gatesGrid, TB, 0, stream>>>(xt, hin, hot,
                                                         Wih_bf, Whh_bf,
                                                         bih, bhh, c_st);
    logits_wmma_kernel<<<logitsGrid, TB, 0, stream>>>(hot, Wout_bf, bout,
                                                      out + (size_t)t * VOCAB);
  }
}
